// ReAttention_25340307047018
// MI455X (gfx1250) — matrix operand order repaired and checked by reading, not compiled, not run on hardware
//
#include <hip/hip_runtime.h>

typedef __attribute__((ext_vector_type(16))) __bf16 bf16x16;
typedef __attribute__((ext_vector_type(8)))  __bf16 bf16x8;
typedef __attribute__((ext_vector_type(8)))  float  floatx8;

#define B_   16
#define N_   576
#define C_   768
#define H_   8
#define HD_  96
#define NT_  36                  // N_/16
#define NN_  (N_ * N_)           // 331776
#define EPS_ 1e-5f
#define SCALE_ 0.10206207261596577f   // 96^-0.5
#define RED_BLOCKS 1024

// ---- WMMA helpers (CDNA5 16x16x32 bf16, f32 accumulate) --------------------
__device__ __forceinline__ bf16x16 load_a_frag(const __bf16* p) {
    // A 16x32 bf16: lane row = caller; elems 0..15 = K kh..kh+15 (contiguous)
    return *(const bf16x16*)p;
}
__device__ __forceinline__ floatx8 wmma_bf(bf16x16 a, bf16x16 b, floatx8 c) {
    return __builtin_amdgcn_wmma_f32_16x16x32_bf16(false, a, false, b, (short)0, c,
                                                   false, false);
}

// ---- Kernel 1: per-token 3x3 convs -> q,k (B,H,N,HD) bf16 ; v^T (B,H,HD,N) --
__global__ void __launch_bounds__(256) k_conv_qkv(
    const float* __restrict__ x,
    const float* __restrict__ wq, const float* __restrict__ wk,
    const float* __restrict__ wv,
    __bf16* __restrict__ qb, __bf16* __restrict__ kb, __bf16* __restrict__ vtb) {
    __shared__ float xin[3 * 256];
    __shared__ float swq[81], swk[81], swv[81];
    const int tid = threadIdx.x;
    const int tok = blockIdx.x;            // b*N_ + n
    const int b = tok / N_, n = tok % N_;
    const float* xp = x + (size_t)tok * C_;
#pragma unroll
    for (int i = 0; i < 3; ++i) xin[i * 256 + tid] = xp[i * 256 + tid];
    if (tid < 81) { swq[tid] = wq[tid]; swk[tid] = wk[tid]; swv[tid] = wv[tid]; }
    __syncthreads();
    const int r = tid >> 4, c = tid & 15;
#pragma unroll
    for (int oc = 0; oc < 3; ++oc) {
        float aq = 0.f, ak = 0.f, av = 0.f;
#pragma unroll
        for (int ic = 0; ic < 3; ++ic) {
#pragma unroll
            for (int kr = 0; kr < 3; ++kr) {
                const int rr = r + kr - 1;
                if (rr < 0 || rr > 15) continue;
#pragma unroll
                for (int kc = 0; kc < 3; ++kc) {
                    const int cc = c + kc - 1;
                    if (cc < 0 || cc > 15) continue;
                    const float xv = xin[ic * 256 + rr * 16 + cc];
                    const int wi = ((oc * 3 + ic) * 3 + kr) * 3 + kc;
                    aq += swq[wi] * xv; ak += swk[wi] * xv; av += swv[wi] * xv;
                }
            }
        }
        const int flat = oc * 256 + r * 16 + c;
        const int h = flat / HD_, d = flat % HD_;
        const size_t qk = (((size_t)b * H_ + h) * N_ + n) * HD_ + d;
        qb[qk] = (__bf16)aq;
        kb[qk] = (__bf16)ak;
        vtb[(((size_t)b * H_ + h) * HD_ + d) * N_ + n] = (__bf16)av;
    }
}

// ---- Kernel 2: S = q k^T (WMMA), row softmax in LDS, write bf16 attn -------
__global__ void __launch_bounds__(256) k_attn_softmax(
    const __bf16* __restrict__ qb, const __bf16* __restrict__ kb,
    __bf16* __restrict__ attn_bf) {
    __shared__ float S[16][580];
    __shared__ float red[16][16];
    __shared__ float rowv[16];
    const int tid = threadIdx.x;
    const int lane = tid & 31, w = tid >> 5;
    int bid = blockIdx.x;
    const int nt = bid % NT_; bid /= NT_;
    const int h = bid % H_;   const int b = bid / H_;
    const int n0 = nt * 16;
    const __bf16* qp = qb + (size_t)(b * H_ + h) * N_ * HD_;
    const __bf16* kp = kb + (size_t)(b * H_ + h) * N_ * HD_;
    const int arow = n0 + (lane & 15);
    const int kh8  = (lane >> 4) << 3;
    const int kh16 = (lane >> 4) << 4;
    const bf16x16 a0 = load_a_frag(qp + (size_t)arow * HD_ +  0 + kh16);
    const bf16x16 a1 = load_a_frag(qp + (size_t)arow * HD_ + 32 + kh16);
    const bf16x16 a2 = load_a_frag(qp + (size_t)arow * HD_ + 64 + kh16);
    for (int mt = w; mt < NT_; mt += 8) {
        const int m0 = mt * 16;
        const __bf16* bp = kp + (size_t)(m0 + (lane & 15)) * HD_ + kh16;
        floatx8 acc = {};
        acc = wmma_bf(a0, *(const bf16x16*)(bp +  0), acc);
        acc = wmma_bf(a1, *(const bf16x16*)(bp + 32), acc);
        acc = wmma_bf(a2, *(const bf16x16*)(bp + 64), acc);
        const int rbase = (lane >> 4) << 3;
        const int col = m0 + (lane & 15);
#pragma unroll
        for (int rr = 0; rr < 8; ++rr) S[rbase + rr][col] = acc[rr];
    }
    __syncthreads();
    // softmax over 576 cols per row; 16 threads per row
    const int row = tid >> 4, l16 = tid & 15;
    float mx = -3.4e38f;
    for (int cc = l16; cc < N_; cc += 16) mx = fmaxf(mx, S[row][cc]);
    red[row][l16] = mx;
    __syncthreads();
    if (l16 == 0) {
        float m2 = red[row][0];
#pragma unroll
        for (int i = 1; i < 16; ++i) m2 = fmaxf(m2, red[row][i]);
        rowv[row] = m2;
    }
    __syncthreads();
    const float rm = rowv[row];
    float sum = 0.f;
    for (int cc = l16; cc < N_; cc += 16) {
        const float e = __expf((S[row][cc] - rm) * SCALE_);
        S[row][cc] = e; sum += e;
    }
    __syncthreads();
    red[row][l16] = sum;
    __syncthreads();
    if (l16 == 0) {
        float s2 = 0.f;
#pragma unroll
        for (int i = 0; i < 16; ++i) s2 += red[row][i];
        rowv[row] = 1.f / s2;
    }
    __syncthreads();
    const float inv = rowv[row];
    __bf16* op = attn_bf + (size_t)(b * H_ + h) * NN_ + (size_t)(n0 + row) * N_;
    for (int cc = l16; cc < N_; cc += 16) op[cc] = (__bf16)(S[row][cc] * inv);
}

// ---- Kernel 3: deterministic partial sums/sumsq of re-attention output -----
__global__ void __launch_bounds__(256) k_stats(
    const __bf16* __restrict__ attn_bf,
    const float* __restrict__ rw, const float* __restrict__ rb,
    float* __restrict__ partials) {
    __shared__ float sw[64]; __shared__ float sb[8];
    __shared__ float rbuf[256];
    const int tid = threadIdx.x;
    if (tid < 64) sw[tid] = rw[tid];
    if (tid < 8)  sb[tid] = rb[tid];
    __syncthreads();
    float ls[8] = {0,0,0,0,0,0,0,0};
    float lq[8] = {0,0,0,0,0,0,0,0};
    const long total = (long)B_ * NN_;
    for (long idx = (long)blockIdx.x * 256 + tid; idx < total;
         idx += (long)gridDim.x * 256) {
        const long b = idx / NN_, off = idx % NN_;
        const __bf16* p = attn_bf + b * (long)H_ * NN_ + off;
        float ph[8];
#pragma unroll
        for (int hh = 0; hh < 8; ++hh) ph[hh] = (float)p[(long)hh * NN_];
#pragma unroll
        for (int g = 0; g < 8; ++g) {
            float a2 = sb[g];
#pragma unroll
            for (int hh = 0; hh < 8; ++hh) a2 += sw[g * 8 + hh] * ph[hh];
            ls[g] += a2; lq[g] += a2 * a2;
        }
    }
#pragma unroll
    for (int i = 0; i < 16; ++i) {
        const float v = (i < 8) ? ls[i] : lq[i - 8];
        __syncthreads();
        rbuf[tid] = v;
        __syncthreads();
        for (int s = 128; s > 0; s >>= 1) {
            if (tid < s) rbuf[tid] += rbuf[tid + s];
            __syncthreads();
        }
        if (tid == 0) partials[(long)blockIdx.x * 16 + i] = rbuf[0];
    }
}

// ---- Kernel 4a: finalize BN scale/shift ------------------------------------
__global__ void __launch_bounds__(256) k_finalize(
    const float* __restrict__ partials,
    const float* __restrict__ gamma, const float* __restrict__ beta,
    float* __restrict__ stats) {
    __shared__ float rbuf[256];
    __shared__ float tot[16];
    const int tid = threadIdx.x;
    for (int i = 0; i < 16; ++i) {
        float v = 0.f;
        for (int j = tid; j < RED_BLOCKS; j += 256) v += partials[j * 16 + i];
        __syncthreads();
        rbuf[tid] = v;
        __syncthreads();
        for (int s = 128; s > 0; s >>= 1) {
            if (tid < s) rbuf[tid] += rbuf[tid + s];
            __syncthreads();
        }
        if (tid == 0) tot[i] = rbuf[0];
        __syncthreads();
    }
    if (tid < 8) {
        const float cnt  = (float)((long)B_ * NN_);
        const float mean = tot[tid] / cnt;
        const float var  = tot[tid + 8] / cnt - mean * mean;
        const float rstd = rsqrtf(var + EPS_);
        const float sc   = gamma[tid] * rstd;
        stats[tid]     = sc;
        stats[tid + 8] = beta[tid] - mean * sc;
    }
}

// ---- Kernel 4b: re-attention mix + BN; fp32 out; bf16 in-place for GEMM ----
__global__ void __launch_bounds__(256) k_reatten_apply(
    __bf16* __restrict__ attn_bf,
    const float* __restrict__ rw, const float* __restrict__ rb,
    const float* __restrict__ stats,
    float* __restrict__ attn_out) {
    __shared__ float sw[64], sb[8], ssc[8], ssh[8];
    const int tid = threadIdx.x;
    if (tid < 64) sw[tid] = rw[tid];
    if (tid < 8) { sb[tid] = rb[tid]; ssc[tid] = stats[tid]; ssh[tid] = stats[tid + 8]; }
    __syncthreads();
    const long total = (long)B_ * NN_;
    const long idx = (long)blockIdx.x * 256 + tid;
    if (idx >= total) return;
    const long b = idx / NN_, off = idx % NN_;
    __bf16* p  = attn_bf + b * (long)H_ * NN_ + off;
    float* op  = attn_out + b * (long)H_ * NN_ + off;
    float ph[8];
#pragma unroll
    for (int hh = 0; hh < 8; ++hh) ph[hh] = (float)p[(long)hh * NN_];
#pragma unroll
    for (int g = 0; g < 8; ++g) {
        float a2 = sb[g];
#pragma unroll
        for (int hh = 0; hh < 8; ++hh) a2 += sw[g * 8 + hh] * ph[hh];
        const float y = a2 * ssc[g] + ssh[g];
        op[(long)g * NN_] = y;               // returned attn (fp32)
        p[(long)g * NN_]  = (__bf16)y;       // feed next GEMM (bf16, in place)
    }
}

// ---- Kernel 5: out = attn2 @ v  (WMMA), write token-major bf16 x2 ----------
__global__ void __launch_bounds__(128) k_attn_v(
    const __bf16* __restrict__ attn2, const __bf16* __restrict__ vtb,
    __bf16* __restrict__ x2) {
    const int tid = threadIdx.x;
    const int lane = tid & 31;
    const int wid = blockIdx.x * 4 + (tid >> 5);      // 0..4607
    const int nt = wid % NT_;
    const int h  = (wid / NT_) % H_;
    const int b  = wid / (NT_ * H_);
    const int n0 = nt * 16;
    const __bf16* Ap = attn2 + (size_t)(b * H_ + h) * NN_;
    const __bf16* Bp = vtb + (size_t)(b * H_ + h) * HD_ * N_;
    const int arow = n0 + (lane & 15);
    const int kh8  = (lane >> 4) << 3;
    const int kh16 = (lane >> 4) << 4;
    floatx8 acc[6];
    const floatx8 z = {};
#pragma unroll
    for (int j = 0; j < 6; ++j) acc[j] = z;
    for (int kc = 0; kc < 18; ++kc) {
        const int k0 = kc * 32;
        const bf16x16 a = load_a_frag(Ap + (size_t)arow * N_ + k0 + kh16);
#pragma unroll
        for (int dt = 0; dt < 6; ++dt) {
            const __bf16* bp = Bp + (size_t)(dt * 16 + (lane & 15)) * N_ + k0 + kh16;
            acc[dt] = wmma_bf(a, *(const bf16x16*)bp, acc[dt]);
        }
    }
    const int rbase = (lane >> 4) << 3;
#pragma unroll
    for (int dt = 0; dt < 6; ++dt) {
        const int d = dt * 16 + (lane & 15);
#pragma unroll
        for (int rr = 0; rr < 8; ++rr) {
            const int n = n0 + rbase + rr;
            x2[((size_t)b * N_ + n) * C_ + h * HD_ + d] = (__bf16)acc[dt][rr];
        }
    }
}

// ---- Kernel 6: projection out = x2 @ proj_w^T + b (WMMA) -------------------
__global__ void __launch_bounds__(256) k_proj(
    const __bf16* __restrict__ x2, const __bf16* __restrict__ pw,
    const float* __restrict__ pbias, float* __restrict__ out) {
    const int tid = threadIdx.x;
    const int lane = tid & 31, w = tid >> 5;
    const int r0 = blockIdx.x * 16;                   // row tile of M = B*N
    const int arow = r0 + (lane & 15);
    const int kh8  = (lane >> 4) << 3;
    const int kh16 = (lane >> 4) << 4;
    floatx8 acc[6];
    const floatx8 z = {};
#pragma unroll
    for (int j = 0; j < 6; ++j) acc[j] = z;
    for (int kc = 0; kc < 24; ++kc) {
        const int k0 = kc * 32;
        const bf16x16 a = load_a_frag(x2 + (size_t)arow * C_ + k0 + kh16);
#pragma unroll
        for (int j = 0; j < 6; ++j) {
            const int c = (w * 6 + j) * 16 + (lane & 15);
            const __bf16* bp = pw + (size_t)c * C_ + k0 + kh16;
            acc[j] = wmma_bf(a, *(const bf16x16*)bp, acc[j]);
        }
    }
    const int rbase = (lane >> 4) << 3;
#pragma unroll
    for (int j = 0; j < 6; ++j) {
        const int c = (w * 6 + j) * 16 + (lane & 15);
        const float bv = pbias[c];
#pragma unroll
        for (int rr = 0; rr < 8; ++rr)
            out[(size_t)(r0 + rbase + rr) * C_ + c] = acc[j][rr] + bv;
    }
}

// ---- tiny fp32 -> bf16 convert ---------------------------------------------
__global__ void __launch_bounds__(256) k_cvt_bf16(
    const float* __restrict__ src, __bf16* __restrict__ dst, int n) {
    const int i = blockIdx.x * 256 + threadIdx.x;
    if (i < n) dst[i] = (__bf16)src[i];
}

extern "C" void kernel_launch(void* const* d_in, const int* in_sizes, int n_in,
                              void* d_out, int out_size, void* d_ws, size_t ws_size,
                              hipStream_t stream) {
    const float* x      = (const float*)d_in[0];
    const float* qw     = (const float*)d_in[1];
    const float* kw     = (const float*)d_in[2];
    const float* vw     = (const float*)d_in[3];
    const float* rw     = (const float*)d_in[4];
    const float* rb     = (const float*)d_in[5];
    const float* gamma  = (const float*)d_in[6];
    const float* beta   = (const float*)d_in[7];
    const float* projw  = (const float*)d_in[8];
    const float* projb  = (const float*)d_in[9];

    char* ws = (char*)d_ws;
    size_t off = 0;
    auto alloc = [&](size_t bytes) -> char* {
        char* p = ws + off;
        off += (bytes + 255) & ~(size_t)255;
        return p;
    };
    const size_t qkv_elems = (size_t)B_ * H_ * N_ * HD_;     // 7,077,888
    __bf16* qb       = (__bf16*)alloc(qkv_elems * 2);
    __bf16* kb       = (__bf16*)alloc(qkv_elems * 2);
    __bf16* vtb      = (__bf16*)alloc(qkv_elems * 2);        // (B,H,HD,N)
    __bf16* attn_bf  = (__bf16*)alloc((size_t)B_ * H_ * NN_ * 2);
    __bf16* x2       = (__bf16*)alloc((size_t)B_ * N_ * C_ * 2);
    __bf16* pw       = (__bf16*)alloc((size_t)C_ * C_ * 2);
    float*  partials = (float*)alloc((size_t)RED_BLOCKS * 16 * 4);
    float*  stats    = (float*)alloc(64);

    float* out      = (float*)d_out;
    float* attn_out = out + (size_t)B_ * N_ * C_;   // outputs concatenated (out, attn)

    k_cvt_bf16<<<(C_ * C_ + 255) / 256, 256, 0, stream>>>(projw, pw, C_ * C_);
    k_conv_qkv<<<B_ * N_, 256, 0, stream>>>(x, qw, kw, vw, qb, kb, vtb);
    k_attn_softmax<<<B_ * H_ * NT_, 256, 0, stream>>>(qb, kb, attn_bf);
    k_stats<<<RED_BLOCKS, 256, 0, stream>>>(attn_bf, rw, rb, partials);
    k_finalize<<<1, 256, 0, stream>>>(partials, gamma, beta, stats);
    k_reatten_apply<<<(int)(((long)B_ * NN_ + 255) / 256), 256, 0, stream>>>(
        attn_bf, rw, rb, stats, attn_out);
    k_attn_v<<<(B_ * H_ * NT_) / 4, 128, 0, stream>>>(attn_bf, vtb, x2);
    k_proj<<<(B_ * N_) / 16, 256, 0, stream>>>(x2, pw, projb, out);
}
